// StablelmAttention_8126078124762
// MI455X (gfx1250) — matrix operand order repaired and checked by reading, not compiled, not run on hardware
//
#include <hip/hip_runtime.h>

// ---------------------------------------------------------------------------
// StableLM attention block for MI455X (gfx1250), bf16 WMMA pipeline.
// GEMM tiles are moved by the Tensor Data Mover (tensor_load_to_lds,
// TENSORcnt) with LDS row padding done by the TDM pad feature; fallback is
// async global->LDS b128 copies (ASYNCcnt).
//   S=2048, E=5120, H=32, KVH=8 (G=4), D=160, ROT=40, QKV_OUT=7680
// ---------------------------------------------------------------------------

typedef __attribute__((ext_vector_type(16))) __bf16       v16bf;
typedef __attribute__((ext_vector_type(8)))  float        v8f;
typedef __attribute__((ext_vector_type(4)))  unsigned int u32x4;
typedef __attribute__((ext_vector_type(4)))  unsigned int uint32x4;
typedef __attribute__((ext_vector_type(8)))  int          int32x8;
typedef __attribute__((ext_vector_type(4)))  int          int32x4;

typedef __attribute__((address_space(3))) void* las_ptr;   // LDS (32-bit)

#if __has_builtin(__builtin_amdgcn_tensor_load_to_lds)
#define USE_TDM 1
#else
#define USE_TDM 0
#endif

union Frag {
    u32x4 q[2];   // two 16-byte LDS loads
    v16bf v;      // WMMA operand view
};

static __device__ inline v8f wmma_bf16(const Frag& a, const Frag& b, v8f c) {
    return __builtin_amdgcn_wmma_f32_16x16x32_bf16(
        /*neg_a=*/false, a.v, /*neg_b=*/false, b.v,
        /*c_mod=*/(short)0, c, /*reuse_a=*/false, /*reuse_b=*/false);
}

static __device__ inline unsigned short f2bf(float x) {
    unsigned u = __builtin_bit_cast(unsigned, x);
    u += 0x7FFFu + ((u >> 16) & 1u);          // round-to-nearest-even
    return (unsigned short)(u >> 16);
}

static __device__ inline ushort4 cvt4(float4 v) {
    ushort4 r;
    r.x = f2bf(v.x); r.y = f2bf(v.y); r.z = f2bf(v.z); r.w = f2bf(v.w);
    return r;
}

// --- fallback: 16-byte global -> LDS copy via GLOBAL_LOAD_ASYNC_TO_LDS_B128
static __device__ inline void async_copy_b128(const void* g, void* l) {
#if defined(__gfx1250__)
    unsigned lds = (unsigned)(size_t)(las_ptr)l;   // AS(3) ptr == LDS byte offset
    asm volatile("global_load_async_to_lds_b128 %0, %1, off"
                 :: "v"(lds), "v"(g) : "memory");
#else
    *(u32x4*)l = *(const u32x4*)g;
#endif
}

static __device__ inline void wait_async() {
#if defined(__gfx1250__)
    asm volatile("s_wait_asynccnt 0" ::: "memory");
#endif
}

static __device__ inline void wait_tensor() {
#if __has_builtin(__builtin_amdgcn_s_wait_tensorcnt)
    __builtin_amdgcn_s_wait_tensorcnt(0);
#elif defined(__gfx1250__)
    asm volatile("s_wait_tensorcnt 0" ::: "memory");
#endif
}

#if USE_TDM
// TDM 2D tile load: bf16 tensor, tile t0 (contiguous) x t1 rows, row stride
// d0_stride elements. LDS pad: after every 16 DWORDs (one 32-elem row, 64 B)
// insert 4 DWORDs (16 B) -> LDS row stride 80 B (= GLDS ushorts).
// D# layout per CDNA5 ISA ch.8 (group0: 128b, group1: 256b; groups 2/3/4 zero).
static __device__ inline void tdm_load_tile_bf16(
    const unsigned short* g, unsigned lds_byte_addr,
    int d0_stride, int t0, int t1)
{
    unsigned long long ga = (unsigned long long)(size_t)g;
    uint32x4 g0 = {
        1u,                                               // count=1 (valid user D#)
        lds_byte_addr,                                    // lds_addr [63:32]
        (unsigned)ga,                                     // global_addr [95:64]
        ((unsigned)(ga >> 32) & 0x01FFFFFFu) | 0x80000000u// addr[56:32] | type=2
    };
    int32x8 g1;
    g1[0] = (1 << 16)            // data_size = 1 (2 bytes)
          | (1 << 20)            // pad_enable
          | (3 << 22)            // pad_interval: 16 DWORDs
          | (3 << 25);           // pad_amount:   4 DWORDs
    g1[1] = (t0 & 0xFFFF) << 16;                          // tensor_dim0 = t0
    g1[2] = (t1 & 0xFFFF) << 16;                          // tensor_dim1 = t1
    g1[3] = (t0 & 0xFFFF) << 16;                          // tile_dim0 = t0
    g1[4] = (t1 & 0xFFFF);                                // tile_dim1 = t1 (tile_dim2=0)
    g1[5] = d0_stride;                                    // tensor_dim0_stride[31:0]
    g1[6] = 0;                                            // stride[47:32] | dim1_stride lo
    g1[7] = 0;
    int32x4 z4 = { 0, 0, 0, 0 };
    int32x8 z8 = { 0, 0, 0, 0, 0, 0, 0, 0 };
    __builtin_amdgcn_tensor_load_to_lds(g0, g1, z4, z4, z8, 0);
}
#endif

// ---------------------------------------------------------------------------
// f32 -> bf16 bulk conversion (grid-stride, float4 vectorized; n % 4 == 0)
// ---------------------------------------------------------------------------
__global__ __launch_bounds__(256) void cvt_kernel(
    const float* __restrict__ src, unsigned short* __restrict__ dst, int n4)
{
    int i = blockIdx.x * blockDim.x + threadIdx.x;
    const int stride = gridDim.x * blockDim.x;
    for (; i < n4; i += stride)
        ((ushort4*)dst)[i] = cvt4(((const float4*)src)[i]);
}

// ---------------------------------------------------------------------------
// GEMM: C[M,N] = A[M,K] * B[N,K]^T (+ bias[N]); A and B are bf16, K-contig.
// Block = 256 threads = 8 waves; tile 128(M) x 128(N) x 32(K), double-
// buffered LDS. Tiles moved by TDM (wave 0 -> A, wave 1 -> B), one
// tensor_load_to_lds per matrix per k-step. Each wave owns 32x64 = 2x4 WMMAs.
// ---------------------------------------------------------------------------
#define GLDS 40   // padded LDS row stride in ushorts (80 B, 16B-aligned rows)

__global__ __launch_bounds__(256) void gemm_bf16(
    const unsigned short* __restrict__ A, const unsigned short* __restrict__ B,
    const float* __restrict__ bias, float* __restrict__ C,
    int M, int N, int K)
{
    __shared__ unsigned short As[2][128 * GLDS];
    __shared__ unsigned short Bs[2][128 * GLDS];

    const int tid  = threadIdx.x;
    const int lane = tid & 31;
    const int wave = tid >> 5;
    const int mblk = blockIdx.y * 128;
    const int nblk = blockIdx.x * 128;
    const int mw   = (wave & 3) * 32;   // wave's M offset in tile
    const int nw   = (wave >> 2) * 64;  // wave's N offset in tile

    const int r15 = lane & 15;
    const int hi  = lane >> 4;          // lane half: selects K sub-range
    const int aoff0 = hi ? 8 : 0;       // A frag: K base for VGPRs 0-3
    const int aoff1 = hi ? 24 : 16;     // A frag: K base for VGPRs 4-7
    const int boff0 = hi ? 8 : 0;       // B frag: K base for VGPRs 0-3 (+16 for 4-7)

    v8f acc[2][4] = {};

#if USE_TDM
    const unsigned ldsA0 = (unsigned)(size_t)(las_ptr)&As[0][0];
    const unsigned ldsB0 = (unsigned)(size_t)(las_ptr)&Bs[0][0];
    const unsigned bufBytes = 128 * GLDS * 2;
    const unsigned short* Abase = A + (size_t)mblk * K;
    const unsigned short* Bbase = B + (size_t)nblk * K;

    auto fill = [&](int buf, int k0) {
        if (wave == 0)
            tdm_load_tile_bf16(Abase + k0, ldsA0 + buf * bufBytes, K, 32, 128);
        else if (wave == 1)
            tdm_load_tile_bf16(Bbase + k0, ldsB0 + buf * bufBytes, K, 32, 128);
    };
    auto wait_fill = [&]() { if (wave < 2) wait_tensor(); };
#else
    // per-thread async fill: each thread copies 32 B of A and B (2x b128 each)
    const int lrow = tid >> 1;          // 0..127: tile row
    const int lk   = (tid & 1) * 16;    // 16-ushort half of 32-wide K slab
    const unsigned short* Ag = A + (size_t)(mblk + lrow) * K + lk;
    const unsigned short* Bg = B + (size_t)(nblk + lrow) * K + lk;

    auto fill = [&](int buf, int k0) {
        unsigned short* al = &As[buf][lrow * GLDS + lk];
        unsigned short* bl = &Bs[buf][lrow * GLDS + lk];
        async_copy_b128(Ag + k0,     al);
        async_copy_b128(Ag + k0 + 8, al + 8);
        async_copy_b128(Bg + k0,     bl);
        async_copy_b128(Bg + k0 + 8, bl + 8);
    };
    auto wait_fill = [&]() { wait_async(); };
#endif

    fill(0, 0);
    wait_fill();
    __syncthreads();

    const int nsteps = K / 32;
    for (int s = 0; s < nsteps; ++s) {
        const int p = s & 1;
        if (s + 1 < nsteps) fill(p ^ 1, (s + 1) * 32);

        Frag af[2];
        #pragma unroll
        for (int t = 0; t < 2; ++t) {
            const unsigned short* arow = &As[p][(mw + 16 * t + r15) * GLDS];
            af[t].q[0] = *(const u32x4*)(arow + aoff0);
            af[t].q[1] = *(const u32x4*)(arow + aoff1);
        }
        #pragma unroll
        for (int n = 0; n < 4; ++n) {
            Frag bf;
            const unsigned short* brow = &Bs[p][(nw + 16 * n + r15) * GLDS];
            bf.q[0] = *(const u32x4*)(brow + boff0);
            bf.q[1] = *(const u32x4*)(brow + boff0 + 16);
            #pragma unroll
            for (int t = 0; t < 2; ++t)
                acc[t][n] = wmma_bf16(af[t], bf, acc[t][n]);
        }
        wait_fill();
        __syncthreads();
    }

    // epilogue: C layout — VGPR i: row = i + 8*hi, col = lane&15
    #pragma unroll
    for (int n = 0; n < 4; ++n) {
        const int col = nblk + nw + 16 * n + r15;
        const float bv = bias ? bias[col] : 0.0f;
        #pragma unroll
        for (int t = 0; t < 2; ++t) {
            #pragma unroll
            for (int i = 0; i < 8; ++i) {
                const int row = mblk + mw + 16 * t + hi * 8 + i;
                C[(size_t)row * N + col] = acc[t][n][i] + bv;
            }
        }
    }
}

// ---------------------------------------------------------------------------
// Partial RoPE in-place on qkv[2048][7680] (f32): rotate first 40 dims of
// the 32 q heads and 8 k heads.
// ---------------------------------------------------------------------------
__global__ __launch_bounds__(256) void rope_kernel(
    float* __restrict__ qkv, const int* __restrict__ positions)
{
    const int s = blockIdx.x;
    const float pos = (float)positions[s];
    float* row = qkv + (size_t)s * 7680;

    for (int item = threadIdx.x; item < 40 * 20; item += blockDim.x) {
        const int hidx = item / 20;
        const int i    = item % 20;
        const int col  = (hidx < 32) ? hidx * 160 : 5120 + (hidx - 32) * 160;
        const float inv_freq = __powf(10000.0f, -(float)(2 * i) / 40.0f);
        float sn, cs;
        __sincosf(pos * inv_freq, &sn, &cs);
        const float x1 = row[col + i];
        const float x2 = row[col + 20 + i];
        row[col + i]      = x1 * cs - x2 * sn;
        row[col + 20 + i] = x2 * cs + x1 * sn;
    }
}

// ---------------------------------------------------------------------------
// Flash attention, one (head, 64-row q-tile) per 128-thread block (4 waves).
// Each wave owns 16 q rows. D=160 contraction = 5 bf16-WMMA K-steps; 64-key
// tiles; online softmax state in LDS; P overwrites the score buffer in place
// as bf16; K-tile and V^T-tile share one LDS buffer. Output written as bf16
// (it is the A operand of the O-projection GEMM).
// ---------------------------------------------------------------------------
#define QLDS 168   // Q/K LDS row stride (ushorts): 160 + pad, 16B-aligned
#define VLDS 72    // V^T LDS row stride (ushorts): 64 + pad, 16B-aligned
#define SLDS 68    // score row stride (f32): rows 272 B, 16B-aligned

__global__ __launch_bounds__(128) void attn_kernel(
    const float* __restrict__ qkv, unsigned short* __restrict__ out)
{
    __shared__ unsigned short Qs[64 * QLDS];       // 21504 B
    __shared__ unsigned short KV[160 * VLDS];      // 23040 B: Ks[64][QLDS] | Vt[160][VLDS]
    __shared__ float          Sb[64 * SLDS];       // 17408 B: scores f32, then P bf16
    __shared__ float mBuf[64], lBuf[64], aBuf[64];

    const int tid  = threadIdx.x;
    const int lane = tid & 31;
    const int wave = tid >> 5;
    const int qt   = blockIdx.x;        // q tile (0..31)
    const int h    = blockIdx.y;        // q head (0..31)
    const int kvh  = h >> 2;            // G = 4
    const int qcol = h * 160;
    const int kcol = 5120 + kvh * 160;
    const int vcol = 6400 + kvh * 160;
    const int qrow0 = qt * 64;
    const int r15 = lane & 15;
    const int hi  = lane >> 4;
    const int qb  = wave * 16;          // wave's q-row base in tile
    const float scale = 0.07905694150420949f;   // 160^-0.5

    // load Q tile (64 x 160 f32 -> bf16)
    for (int j = tid; j < 64 * 40; j += 128) {
        const int r  = j / 40;
        const int c4 = (j % 40) * 4;
        float4 v = *(const float4*)(qkv + (size_t)(qrow0 + r) * 7680 + qcol + c4);
        *(ushort4*)&Qs[r * QLDS + c4] = cvt4(v);
    }
    if (tid < 64) { mBuf[tid] = -3e38f; lBuf[tid] = 0.0f; }

    v8f oacc[10] = {};
    __syncthreads();

    for (int kt = 0; kt <= qt; ++kt) {
        const int kr0 = kt * 64;

        // K tile -> KV viewed as Ks[64][QLDS]
        for (int j = tid; j < 64 * 40; j += 128) {
            const int r  = j / 40;
            const int c4 = (j % 40) * 4;
            float4 v = *(const float4*)(qkv + (size_t)(kr0 + r) * 7680 + kcol + c4);
            *(ushort4*)&KV[r * QLDS + c4] = cvt4(v);
        }
        __syncthreads();

        // scores: S = Q . K^T   (16 x 64 per wave)
        v8f sacc[4] = {};
        #pragma unroll
        for (int d = 0; d < 5; ++d) {
            Frag qa;
            const unsigned short* qrow = &Qs[(qb + r15) * QLDS + d * 32];
            qa.q[0] = *(const u32x4*)(qrow + (hi ? 8 : 0));
            qa.q[1] = *(const u32x4*)(qrow + (hi ? 24 : 16));
            #pragma unroll
            for (int n = 0; n < 4; ++n) {
                Frag kb;
                const unsigned short* krow = &KV[(16 * n + r15) * QLDS + d * 32 + (hi ? 8 : 0)];
                kb.q[0] = *(const u32x4*)krow;
                kb.q[1] = *(const u32x4*)(krow + 16);
                sacc[n] = wmma_bf16(qa, kb, sacc[n]);
            }
        }
        #pragma unroll
        for (int n = 0; n < 4; ++n)
            #pragma unroll
            for (int i = 0; i < 8; ++i)
                Sb[(qb + hi * 8 + i) * SLDS + 16 * n + r15] = sacc[n][i];
        __syncthreads();

        // V tile -> KV as Vt[160][VLDS] (transposed; Ks reads done)
        for (int j = tid; j < 64 * 40; j += 128) {
            const int r  = j / 40;
            const int c4 = (j % 40) * 4;
            float4 v = *(const float4*)(qkv + (size_t)(kr0 + r) * 7680 + vcol + c4);
            KV[(c4 + 0) * VLDS + r] = f2bf(v.x);
            KV[(c4 + 1) * VLDS + r] = f2bf(v.y);
            KV[(c4 + 2) * VLDS + r] = f2bf(v.z);
            KV[(c4 + 3) * VLDS + r] = f2bf(v.w);
        }

        // online softmax: one thread per q row; P overwrites S in place (bf16)
        if (tid < 64) {
            const int row  = tid;
            const int lim  = (kt == qt) ? (row + 1) : 64;   // causal valid cols
            float* srow = &Sb[row * SLDS];
            const float mOld = mBuf[row];
            float rmax = -3e38f;
            for (int c = 0; c < lim; ++c)
                rmax = fmaxf(rmax, srow[c] * scale);
            const float mNew  = fmaxf(mOld, rmax);
            const float alpha = __expf(mOld - mNew);
            unsigned short* prow = (unsigned short*)srow;
            float lsum = 0.0f;
            for (int c = 0; c < 64; ++c) {
                const float p = (c < lim) ? __expf(srow[c] * scale - mNew) : 0.0f;
                lsum += p;
                prow[c] = f2bf(p);      // writes trail reads: safe in place
            }
            mBuf[row] = mNew;
            lBuf[row] = alpha * lBuf[row] + lsum;
            aBuf[row] = alpha;
        }
        __syncthreads();

        // rescale O and accumulate O += P . V
        float af[8];
        #pragma unroll
        for (int i = 0; i < 8; ++i) af[i] = aBuf[qb + hi * 8 + i];
        #pragma unroll
        for (int nt = 0; nt < 10; ++nt)
            #pragma unroll
            for (int i = 0; i < 8; ++i)
                oacc[nt][i] *= af[i];

        #pragma unroll
        for (int kk = 0; kk < 2; ++kk) {
            Frag pa;
            const unsigned short* prow =
                (const unsigned short*)&Sb[(qb + r15) * SLDS] + kk * 32;
            pa.q[0] = *(const u32x4*)(prow + (hi ? 8 : 0));
            pa.q[1] = *(const u32x4*)(prow + (hi ? 24 : 16));
            #pragma unroll
            for (int nt = 0; nt < 10; ++nt) {
                Frag vb;
                const unsigned short* vrow =
                    &KV[(16 * nt + r15) * VLDS + kk * 32 + (hi ? 8 : 0)];
                vb.q[0] = *(const u32x4*)vrow;
                vb.q[1] = *(const u32x4*)(vrow + 16);
                oacc[nt] = wmma_bf16(pa, vb, oacc[nt]);
            }
        }
        __syncthreads();
    }

    // normalize and store bf16: out[s][h*160 + d]
    float linv[8];
    #pragma unroll
    for (int i = 0; i < 8; ++i) linv[i] = 1.0f / lBuf[qb + hi * 8 + i];
    #pragma unroll
    for (int nt = 0; nt < 10; ++nt) {
        const int d = 16 * nt + r15;
        #pragma unroll
        for (int i = 0; i < 8; ++i) {
            const int row = qrow0 + qb + hi * 8 + i;
            out[(size_t)row * 5120 + qcol + d] = f2bf(oacc[nt][i] * linv[i]);
        }
    }
}

// ---------------------------------------------------------------------------
extern "C" void kernel_launch(void* const* d_in, const int* in_sizes, int n_in,
                              void* d_out, int out_size, void* d_ws, size_t ws_size,
                              hipStream_t stream) {
    const float* hidden    = (const float*)d_in[0];  // [1,2048,5120]
    const int*   positions = (const int*)d_in[1];    // [2048]
    const float* w_qkv     = (const float*)d_in[2];  // [7680,5120]
    const float* b_qkv     = (const float*)d_in[3];  // [7680]
    const float* w_o       = (const float*)d_in[4];  // [5120,5120]
    float* out = (float*)d_out;                      // [2048,5120]

    // workspace layout (all 16B-aligned)
    float*          qkv       = (float*)d_ws;                          // [2048,7680] f32
    unsigned short* hidden_bf = (unsigned short*)(qkv + (size_t)2048 * 7680);
    unsigned short* wqkv_bf   = hidden_bf + (size_t)2048 * 5120;
    unsigned short* wo_bf     = wqkv_bf + (size_t)7680 * 5120;
    unsigned short* attn_bf   = wo_bf + (size_t)5120 * 5120;           // [2048,5120] bf16

    // 0) one-time f32 -> bf16 conversions
    cvt_kernel<<<1024, 256, 0, stream>>>(hidden, hidden_bf, 2048 * 5120 / 4);
    cvt_kernel<<<2048, 256, 0, stream>>>(w_qkv, wqkv_bf, 7680 * 5120 / 4);
    cvt_kernel<<<2048, 256, 0, stream>>>(w_o, wo_bf, 5120 * 5120 / 4);

    // 1) QKV projection (+bias)
    gemm_bf16<<<dim3(7680 / 128, 2048 / 128), 256, 0, stream>>>(
        hidden_bf, wqkv_bf, b_qkv, qkv, 2048, 7680, 5120);
    // 2) partial RoPE on q & k, in place (f32)
    rope_kernel<<<dim3(2048), 256, 0, stream>>>(qkv, positions);
    // 3) GQA causal flash attention -> bf16
    attn_kernel<<<dim3(32, 32), 128, 0, stream>>>(qkv, attn_bf);
    // 4) output projection
    gemm_bf16<<<dim3(5120 / 128, 2048 / 128), 256, 0, stream>>>(
        attn_bf, wo_bf, nullptr, out, 2048, 5120, 5120);
}
